// myOTLoss_65017214927534
// MI455X (gfx1250) — compile-verified, matrix operand order checked by reading, not run on hardware
//
#include <hip/hip_runtime.h>
#include <hip/hip_bf16.h>
#include <math.h>

// ---------------------------------------------------------------------------
// Sinkhorn OT on MI455X (gfx1250, wave32, WMMA bf16)
//
// bs=64, d1=d2=768, LAM=0.01, 100 iterations.
// Each sinkstep is logsumexp_i(-dist/lam + log_u)  ==  shifted exp-GEMM:
//   S[b,j] = sum_i exp(A[i,j]-colmax[j]) * exp(log_u[b,i]-su[b])
//   lse    = log(S) + colmax[j] + su[b]
// The exp'd kernel matrices are iteration-invariant -> pre-pack once in the
// exact v_wmma_f32_16x16x32_bf16 B-fragment layout; the hot loop is 200
// dependent fused GEMM kernels.
// ---------------------------------------------------------------------------

#define BS   64
#define DIM  768
#define INV_LAM 100.0f          // 1/0.01
#define NITER 100
#define NTILES (DIM / 16)       // 48 n-tiles
#define KTILES (DIM / 32)       // 24 k-steps of 32

typedef __attribute__((ext_vector_type(16))) __bf16 v16bf;
typedef __attribute__((ext_vector_type(8)))  __bf16 v8bf;
typedef __attribute__((ext_vector_type(8)))  float  v8f;

// ---------------- precompute: column stats of A = -dist/lam ----------------
__global__ __launch_bounds__(256)
void colstats_kernel(const float* __restrict__ dist,
                     float* __restrict__ colmax,    // max_i(-d/lam)
                     float* __restrict__ colmaxB) { // max_i(log d - d/lam)
    const int j = blockIdx.x, tid = threadIdx.x;
    __shared__ float sA[256], sB[256];
    float ma = -3.4e38f, mb = -3.4e38f;
    for (int i = tid; i < DIM; i += 256) {
        float d = dist[i * DIM + j];
        float A = -d * INV_LAM;
        ma = fmaxf(ma, A);
        mb = fmaxf(mb, logf(d) + A);
    }
    sA[tid] = ma; sB[tid] = mb; __syncthreads();
    for (int s = 128; s > 0; s >>= 1) {
        if (tid < s) { sA[tid] = fmaxf(sA[tid], sA[tid + s]);
                       sB[tid] = fmaxf(sB[tid], sB[tid + s]); }
        __syncthreads();
    }
    if (tid == 0) { colmax[j] = sA[0]; colmaxB[j] = sB[0]; }
}

__global__ __launch_bounds__(256)
void rowstats_kernel(const float* __restrict__ dist,
                     float* __restrict__ rowmax) {  // max_j(-d/lam)
    const int i = blockIdx.x, tid = threadIdx.x;
    __shared__ float sA[256];
    float ma = -3.4e38f;
    for (int j = tid; j < DIM; j += 256)
        ma = fmaxf(ma, -dist[i * DIM + j] * INV_LAM);
    sA[tid] = ma; __syncthreads();
    for (int s = 128; s > 0; s >>= 1) {
        if (tid < s) sA[tid] = fmaxf(sA[tid], sA[tid + s]);
        __syncthreads();
    }
    if (tid == 0) rowmax[i] = sA[0];
}

// ------------- precompute: logs of inputs + initial potentials -------------
__global__ __launch_bounds__(256)
void loginit_kernel(const float* __restrict__ mu, const float* __restrict__ nu,
                    float* __restrict__ log_mu, float* __restrict__ log_nu,
                    float* __restrict__ log_u) {
    int idx = blockIdx.x * 256 + threadIdx.x;
    if (idx < BS * DIM) {
        log_mu[idx] = logf(mu[idx]);
        log_nu[idx] = logf(nu[idx]);
        log_u[idx]  = -logf((float)DIM);
    }
}

// -------- pack constant kernel matrices into WMMA B-fragment layout --------
// Fragment layout (16-bit B, 32x16): lane L: n = L%16, khalf = L/16;
// element e (0..15): t=e/2, half=e%2, k = (t<4 ? 2t : 16+2(t-4)) + 8*khalf + half
// mode 0: B[k,n] = exp(-dist[k,n]/lam          - shift[n])   (step 1, K=i)
// mode 1: B[k,n] = exp(-dist[n,k]/lam          - shift[n])   (step 2, K=j)
// mode 2: B[k,n] = exp(log d[k,n] - d[k,n]/lam - shift[n])   (final transport)
__global__ __launch_bounds__(32)
void pack_kernel(const float* __restrict__ dist, const float* __restrict__ shift,
                 __bf16* __restrict__ Kp, int mode) {
    const int lane = threadIdx.x;
    const int nt = blockIdx.x;          // 0..47
    const int kt = blockIdx.y;          // 0..23
    const int nloc = lane & 15, khalf = lane >> 4;
    const int n = nt * 16 + nloc;
    const float sh = shift[n];
    v16bf frag;
#pragma unroll
    for (int e = 0; e < 16; ++e) {
        int t = e >> 1, half = e & 1;
        int k = kt * 32 + ((t < 4) ? 2 * t : 16 + 2 * (t - 4)) + 8 * khalf + half;
        float d = (mode == 1) ? dist[n * DIM + k] : dist[k * DIM + n];
        float A = -d * INV_LAM;
        float v = (mode == 2) ? (logf(d) + A) : A;
        frag[e] = (__bf16)expf(v - sh);
    }
    *(v16bf*)(Kp + ((size_t)(nt * KTILES + kt) * 32 + lane) * 16) = frag;
}

// ------------------- fused sinkstep: exp -> GEMM -> log --------------------
// out[b,j] = addend[b,j] + sign * ( log(sum_i exp(logx[b,i]-su[b]) * Kp[i,j])
//                                   + shift[j] + su[b] )
// grid (BS/16, DIM/128), block 256 = 8 waves; wave w owns n-tile (blk.y*8+w).
__global__ __launch_bounds__(256)
void sinkstep_gemm_kernel(const float* __restrict__ logx,
                          const __bf16* __restrict__ Kp,
                          const float* __restrict__ shift,
                          const float* __restrict__ addend,
                          float* __restrict__ out,
                          float sign) {
    const int m0 = blockIdx.x * 16;
    const int n0 = blockIdx.y * 128;
    const int tid  = threadIdx.x;
    const int wave = tid >> 5;
    const int lane = tid & 31;

    __shared__ __align__(32) __bf16 aS[16][DIM + 8];  // +8 bf16 pad: kill bank aliasing
    __shared__ float smax[16][16];
    __shared__ float su[16];

    // pass 1: per-row max of logx over K=768 (16 threads per row, 48 elems each)
    {
        const int row = tid >> 4, chunk = tid & 15;
        const float* rp = logx + (size_t)(m0 + row) * DIM + chunk * 48;
        float pm = -3.4e38f;
#pragma unroll 4
        for (int t = 0; t < 48; ++t) pm = fmaxf(pm, rp[t]);
        smax[row][chunk] = pm;
    }
    __syncthreads();
    if (tid < 16) {
        float pm = smax[tid][0];
#pragma unroll
        for (int c = 1; c < 16; ++c) pm = fmaxf(pm, smax[tid][c]);
        su[tid] = pm;
    }
    __syncthreads();

    // pass 2: aS[r][c] = bf16( exp(logx[m0+r,c] - su[r]) )
    for (int idx = tid; idx < 16 * DIM; idx += 256) {
        int r = idx / DIM, c = idx - r * DIM;
        aS[r][c] = (__bf16)expf(logx[(size_t)(m0 + r) * DIM + c] - su[r]);
    }
    __syncthreads();

    // GEMM: 24 x v_wmma_f32_16x16x32_bf16
    const int m = lane & 15, kh = lane >> 4;
    const __bf16* Bb = Kp + (size_t)(n0 / 16 + wave) * KTILES * 512;  // 32 lanes * 16 elems
    v8f acc = {};
    for (int kt = 0; kt < KTILES; ++kt) {
        v8bf lo = *(const v8bf*)&aS[m][kt * 32 + kh * 8];
        v8bf hi = *(const v8bf*)&aS[m][kt * 32 + 16 + kh * 8];
        v16bf av;
#pragma unroll
        for (int e = 0; e < 8; ++e) { av[e] = lo[e]; av[e + 8] = hi[e]; }
        v16bf bv = *(const v16bf*)(Bb + (size_t)(kt * 32 + lane) * 16);
        acc = __builtin_amdgcn_wmma_f32_16x16x32_bf16(
                  false, av, false, bv, (short)0, acc, false, false);
    }

    // epilogue: D layout -> lane L holds col n0+16w+(L%16), rows r+8*(L/16)
    const int col = n0 + wave * 16 + (lane & 15);
    const float sh = shift[col];
#pragma unroll
    for (int r = 0; r < 8; ++r) {
        int lrow = r + 8 * kh;
        int b = m0 + lrow;
        float lse = logf(fmaxf(acc[r], 1e-38f)) + sh + su[lrow];
        out[(size_t)b * DIM + col] = addend[(size_t)b * DIM + col] + sign * lse;
    }
}

// ------------- final: distances[b] = exp(logsumexp_j R[b,j]) ---------------
__global__ __launch_bounds__(256)
void finalred_kernel(const float* __restrict__ R, float* __restrict__ out) {
    const int b = blockIdx.x, tid = threadIdx.x;
    __shared__ float sm[256];
    float pm = -3.4e38f;
    for (int j = tid; j < DIM; j += 256) pm = fmaxf(pm, R[(size_t)b * DIM + j]);
    sm[tid] = pm; __syncthreads();
    for (int s = 128; s > 0; s >>= 1) {
        if (tid < s) sm[tid] = fmaxf(sm[tid], sm[tid + s]);
        __syncthreads();
    }
    float mx = sm[0]; __syncthreads();
    float ps = 0.0f;
    for (int j = tid; j < DIM; j += 256) ps += expf(R[(size_t)b * DIM + j] - mx);
    sm[tid] = ps; __syncthreads();
    for (int s = 128; s > 0; s >>= 1) {
        if (tid < s) sm[tid] += sm[tid + s];
        __syncthreads();
    }
    if (tid == 0) out[b] = sm[0] * expf(mx);
}

// ---------------------------------------------------------------------------
extern "C" void kernel_launch(void* const* d_in, const int* in_sizes, int n_in,
                              void* d_out, int out_size, void* d_ws, size_t ws_size,
                              hipStream_t stream) {
    const float* mu   = (const float*)d_in[0];  // [64,768]
    const float* nu   = (const float*)d_in[1];  // [64,768]
    const float* dist = (const float*)d_in[2];  // [768,768]
    float* distances  = (float*)d_out;          // [64]

    // workspace carve-up (256B aligned)
    char* w = (char*)d_ws;
    size_t off = 0;
    auto take = [&](size_t bytes) -> void* {
        void* p = w + off;
        off = (off + bytes + 255) & ~(size_t)255;
        return p;
    };
    const size_t kbytes = (size_t)DIM * DIM * sizeof(__bf16);   // 1.125 MB each
    const size_t vbytes = (size_t)BS * DIM * sizeof(float);     // 192 KB each
    __bf16* K1p   = (__bf16*)take(kbytes);   // step 1 kernel (contract i)
    __bf16* K2p   = (__bf16*)take(kbytes);   // step 2 kernel (contract j)
    __bf16* KBp   = (__bf16*)take(kbytes);   // final transport kernel
    float* colmax  = (float*)take(DIM * sizeof(float));
    float* rowmax  = (float*)take(DIM * sizeof(float));
    float* colmaxB = (float*)take(DIM * sizeof(float));
    float* log_mu  = (float*)take(vbytes);
    float* log_nu  = (float*)take(vbytes);
    float* log_u   = (float*)take(vbytes);
    float* log_v   = (float*)take(vbytes);
    float* Rbuf    = (float*)take(vbytes);

    // ---- one-time precompute (constant across iterations) ----
    colstats_kernel<<<DIM, 256, 0, stream>>>(dist, colmax, colmaxB);
    rowstats_kernel<<<DIM, 256, 0, stream>>>(dist, rowmax);
    loginit_kernel<<<(BS * DIM + 255) / 256, 256, 0, stream>>>(mu, nu, log_mu, log_nu, log_u);
    dim3 pgrid(NTILES, KTILES);
    pack_kernel<<<pgrid, 32, 0, stream>>>(dist, colmax,  K1p, 0);
    pack_kernel<<<pgrid, 32, 0, stream>>>(dist, rowmax,  K2p, 1);
    pack_kernel<<<pgrid, 32, 0, stream>>>(dist, colmaxB, KBp, 2);

    // ---- 100 Sinkhorn iterations: 2 fused WMMA GEMMs each ----
    dim3 ggrid(BS / 16, DIM / 128);   // 4 x 6 blocks, 256 threads (8 waves)
    for (int it = 0; it < NITER; ++it) {
        // log_v = log_nu - lse_i(-dist/lam + log_u)
        sinkstep_gemm_kernel<<<ggrid, 256, 0, stream>>>(log_u, K1p, colmax, log_nu, log_v, -1.0f);
        // log_u = log_mu - lse_j(-dist/lam + log_v)
        sinkstep_gemm_kernel<<<ggrid, 256, 0, stream>>>(log_v, K2p, rowmax, log_mu, log_u, -1.0f);
    }

    // ---- final: R[b,j] = log_v + lse_i(log d - d/lam + log_u); out = exp(lse_j R) ----
    sinkstep_gemm_kernel<<<ggrid, 256, 0, stream>>>(log_u, KBp, colmaxB, log_v, Rbuf, +1.0f);
    finalred_kernel<<<BS, 256, 0, stream>>>(Rbuf, distances);
}
